// RNNModel_55465207661121
// MI455X (gfx1250) — matrix operand order repaired and checked by reading, not compiled, not run on hardware
//
#include <hip/hip_runtime.h>

// ---------------- problem constants ----------------
constexpr int BATCH = 128;
constexpr int TSEQ  = 500;
constexpr int XDIM  = 248;   // raw input last-dim (246 feats + ability + diff)
constexpr int INF   = 246;
constexpr int KX    = 288;   // 265 (246+11+8) padded up to multiple of 32
constexpr int H     = 512;
constexpr int G4    = 2048;  // 4*H
constexpr int ODIM  = 123;
constexpr int OPAD  = 128;

#define TANH1 0.76159415595576488f   // tanh(1.0)

typedef _Float16 v8h  __attribute__((ext_vector_type(8)));
typedef _Float16 v16h __attribute__((ext_vector_type(16)));
typedef float    v8f  __attribute__((ext_vector_type(8)));

__device__ __forceinline__ float fsig(float x)  { return 1.0f / (1.0f + __expf(-x)); }
__device__ __forceinline__ float ftanh(float x) { return 2.0f / (1.0f + __expf(-2.0f * x)) - 1.0f; }

__device__ __forceinline__ v8f wmma16(v16h a, v16h b, v8f c) {
  // v_wmma_f32_16x16x32_f16 : D = A(16x32 f16) * B(32x16 f16) + C(16x16 f32)
  return __builtin_amdgcn_wmma_f32_16x16x32_f16(false, a, false, b, (short)0, c, false, false);
}

// A-fragment (16x32 f16). Lanes 0-15 hold row M=lane, K {0..15};
// lanes 16-31 hold row M=lane-16, K {16..31}.
__device__ __forceinline__ v16h load_afrag(const _Float16* __restrict__ row, int kk, int laneHi) {
  v8h a0 = *(const v8h*)(row + kk + laneHi * 16);
  v8h a1 = *(const v8h*)(row + kk + laneHi * 16 + 8);
  v16h r;
#pragma unroll
  for (int i = 0; i < 8; ++i) { r[i] = a0[i]; r[i + 8] = a1[i]; }
  return r;
}

// ---------------- prep: new_x = tanh(concat(feats, onehot(diff,11), onehot(ab,8))) -> f16, K padded to 288
__global__ __launch_bounds__(256) void prep_newx(const float* __restrict__ x, _Float16* __restrict__ nx) {
  size_t i = (size_t)blockIdx.x * blockDim.x + threadIdx.x;
  const size_t total = (size_t)BATCH * TSEQ * KX;
  if (i >= total) return;
  int k = (int)(i % KX);
  size_t bt = i / KX;
  const float* row = x + bt * XDIM;
  float v;
  if (k < INF) {
    v = ftanh(row[k]);
  } else if (k < INF + 11) {                 // one-hot diff (x[...,-1] == row[247])
    int d = (int)row[247];
    v = (k - INF == d) ? TANH1 : 0.0f;
  } else if (k < INF + 11 + 8) {             // one-hot ability (x[...,-2] == row[246])
    int a = (int)row[246];
    v = (k - (INF + 11) == a) ? TANH1 : 0.0f;
  } else {
    v = 0.0f;                                // K padding
  }
  nx[i] = (_Float16)v;
}

// generic fp32 [Rsrc,Ksrc] -> f16 [Rdst,Kdst] with zero padding
__global__ __launch_bounds__(256) void cvt_pad_f16(const float* __restrict__ src, _Float16* __restrict__ dst,
                                                   int Rsrc, int Ksrc, int Rdst, int Kdst) {
  size_t i = (size_t)blockIdx.x * blockDim.x + threadIdx.x;
  if (i >= (size_t)Rdst * Kdst) return;
  int k = (int)(i % Kdst);
  int r = (int)(i / Kdst);
  float v = (r < Rsrc && k < Ksrc) ? src[(size_t)r * Ksrc + k] : 0.0f;
  dst[i] = (_Float16)v;
}

__global__ __launch_bounds__(256) void bias_sum(const float* __restrict__ a, const float* __restrict__ b,
                                                float* __restrict__ o, int n) {
  int i = blockIdx.x * blockDim.x + threadIdx.x;
  if (i < n) o[i] = a[i] + b[i];
}

// ---------------- persistent fused LSTM scan ----------------
// 32 WGs x 256 threads = 256 waves. Wave w: nj = w>>3 (h-column tile, shared by
// the 8 waves of a WG -> identical W addresses hit WGP$), bm = w&7 (batch tile).
// Per step: acc_g = bias_g + new_x_t @ Wih_g^T + h_{t-1} @ Whh_g^T.
// Register budget (1024 VGPRs): only W_hh fragments stay resident
// (16 ks x 4 g x 8 = 512 VGPRs, hoisted by the compiler across the t-loop);
// the x-projection runs as a rolled, double-buffered loop so its W_ih
// fragments are NOT hoisted (frees ~290 VGPRs -> no scratch spills).
__global__ __launch_bounds__(256) void lstm_scan(
    const _Float16* __restrict__ nx,    // [BATCH, TSEQ, KX] f16
    const _Float16* __restrict__ wih,   // [G4, KX] f16
    const _Float16* __restrict__ whh,   // [G4, H]  f16
    const float*    __restrict__ bias,  // [G4] (b_ih + b_hh)
    _Float16*       __restrict__ hbuf,  // [2, BATCH, H] f16 ping-pong (zeroed)
    _Float16*       __restrict__ hs,    // [TSEQ, BATCH, H] f16
    unsigned*       __restrict__ bar)   // zeroed arrive counter
{
  const int wave   = (blockIdx.x * blockDim.x + threadIdx.x) >> 5;
  const int lane   = threadIdx.x & 31;
  const int laneLo = lane & 15;
  const int laneHi = lane >> 4;
  const int nj     = wave >> 3;   // h-column tile 0..31 (uniform within WG)
  const int bm     = wave & 7;    // batch tile 0..7
  const int rowA   = bm * 16 + laneLo;
  const unsigned NWG = gridDim.x;

  // per-gate constants (hoisted out of the time loop)
  float bv[4];
  const _Float16* wihRow[4];
  const _Float16* whhRow[4];
#pragma unroll
  for (int g = 0; g < 4; ++g) {
    const int n = g * H + nj * 16 + laneLo;
    bv[g]     = bias[n];
    wihRow[g] = wih + (size_t)n * KX + laneHi * 16;  // B-frag: 16 contiguous K halves
    whhRow[g] = whh + (size_t)n * H  + laneHi * 16;
  }
  const _Float16* xbase = nx + (size_t)rowA * TSEQ * KX;

  v8f c = {0.f, 0.f, 0.f, 0.f, 0.f, 0.f, 0.f, 0.f};

  for (int t = 0; t < TSEQ; ++t) {
    const _Float16* hprev = hbuf + (size_t)(t & 1) * (BATCH * H);
    _Float16*       hnext = hbuf + (size_t)((t + 1) & 1) * (BATCH * H);

    v8f acc[4];
#pragma unroll
    for (int g = 0; g < 4; ++g)
      acc[g] = (v8f){bv[g], bv[g], bv[g], bv[g], bv[g], bv[g], bv[g], bv[g]};

    const _Float16* xrow = xbase + (size_t)t * KX;
    const _Float16* hrow = hprev + (size_t)rowA * H;
    if (t + 1 < TSEQ) __builtin_prefetch(xbase + (size_t)(t + 1) * KX, 0, 1);  // global_prefetch_b8

    // ---- input projection: rolled loop, A+B double-buffered (W_ih streamed, not hoisted)
    v16h a0 = load_afrag(xrow, 0, laneHi);
    v16h b0[4];
#pragma unroll
    for (int g = 0; g < 4; ++g) b0[g] = *(const v16h*)(wihRow[g]);
#pragma unroll 2
    for (int ks = 0; ks < KX / 32 - 1; ++ks) {
      v16h a1 = load_afrag(xrow, (ks + 1) * 32, laneHi);
      v16h b1[4];
#pragma unroll
      for (int g = 0; g < 4; ++g) b1[g] = *(const v16h*)(wihRow[g] + (ks + 1) * 32);
#pragma unroll
      for (int g = 0; g < 4; ++g) acc[g] = wmma16(a0, b0[g], acc[g]);
      a0 = a1;
#pragma unroll
      for (int g = 0; g < 4; ++g) b0[g] = b1[g];
    }
    {   // last x k-step; prime first h A-frag
      v16h a1 = load_afrag(hrow, 0, laneHi);
#pragma unroll
      for (int g = 0; g < 4; ++g) acc[g] = wmma16(a0, b0[g], acc[g]);
      a0 = a1;
    }

    // ---- recurrent: K = 512, fully unrolled; W_hh fragments stay in VGPRs,
    // A-frag pipelined one k-step ahead (ternary folds at unroll time).
#pragma unroll
    for (int ks = 0; ks < H / 32; ++ks) {
      v16h a1 = (ks + 1 < H / 32) ? load_afrag(hrow, (ks + 1) * 32, laneHi) : a0;
#pragma unroll
      for (int g = 0; g < 4; ++g) {
        v16h b = *(const v16h*)(whhRow[g] + ks * 32);
        acc[g] = wmma16(a0, b, acc[g]);
      }
      a0 = a1;
    }

    // gate math + state update (c held in VGPRs across all steps)
    v8f hv;
#pragma unroll
    for (int e = 0; e < 8; ++e) {
      float ig = fsig(acc[0][e]);
      float fg = fsig(acc[1][e]);
      float gg = ftanh(acc[2][e]);
      float og = fsig(acc[3][e]);
      float cn = fg * c[e] + ig * gg;
      c[e] = cn;
      hv[e] = og * ftanh(cn);
    }
    // scatter D-layout: vgpr e -> (row = bm*16 + e + 8*laneHi, col = nj*16 + laneLo)
#pragma unroll
    for (int e = 0; e < 8; ++e) {
      int r = bm * 16 + e + laneHi * 8;
      int cidx = nj * 16 + laneLo;
      _Float16 hh = (_Float16)hv[e];
      hnext[(size_t)r * H + cidx] = hh;
      hs[((size_t)t * BATCH + r) * H + cidx] = hh;
    }

    // ---- device-wide step barrier: monotonic arrive counter ----
    __syncthreads();
    if (threadIdx.x == 0) {
      __threadfence();  // release h stores to agent scope
      __hip_atomic_fetch_add(bar, 1u, __ATOMIC_RELEASE, __HIP_MEMORY_SCOPE_AGENT);
      const unsigned target = NWG * (unsigned)(t + 1);
      while (__hip_atomic_load(bar, __ATOMIC_ACQUIRE, __HIP_MEMORY_SCOPE_AGENT) < target)
        __builtin_amdgcn_s_sleep(1);
    }
    __syncthreads();
    __threadfence();    // acquire: invalidate stale h lines before next step reads
  }
}

// ---------------- final FC: sigmoid(hs @ Wfc^T + b_fc), drop t=499 ----------------
__global__ __launch_bounds__(256) void fc_out(
    const _Float16* __restrict__ hs,   // [TSEQ*BATCH, H] f16 (row r = t*128 + b)
    const _Float16* __restrict__ wfc,  // [OPAD, H] f16, rows >=123 zero
    const float*    __restrict__ bfc,  // [123]
    float*          __restrict__ out)  // [BATCH, 499, ODIM]
{
  const int wave   = (blockIdx.x * blockDim.x + threadIdx.x) >> 5;  // M-tile 0..3999
  const int lane   = threadIdx.x & 31;
  const int laneLo = lane & 15;
  const int laneHi = lane >> 4;
  const int mt     = wave;
  const int rowA   = mt * 16 + laneLo;

  v8f acc[8];
#pragma unroll
  for (int nt = 0; nt < 8; ++nt) {
    int n = nt * 16 + laneLo;
    float b0 = (n < ODIM) ? bfc[n] : 0.0f;
    acc[nt] = (v8f){b0, b0, b0, b0, b0, b0, b0, b0};
  }

  const _Float16* arow = hs + (size_t)rowA * H;
  v16h a = load_afrag(arow, 0, laneHi);   // pipelined A-frag (ternary folds at unroll time)
#pragma unroll
  for (int ks = 0; ks < H / 32; ++ks) {
    v16h an = (ks + 1 < H / 32) ? load_afrag(arow, (ks + 1) * 32, laneHi) : a;
#pragma unroll
    for (int nt = 0; nt < 8; ++nt) {
      int n = nt * 16 + laneLo;
      v16h b = *(const v16h*)(wfc + (size_t)n * H + ks * 32 + laneHi * 16);
      acc[nt] = wmma16(a, b, acc[nt]);
    }
    a = an;
  }

#pragma unroll
  for (int nt = 0; nt < 8; ++nt) {
    int n = nt * 16 + laneLo;
#pragma unroll
    for (int e = 0; e < 8; ++e) {
      int r = mt * 16 + e + laneHi * 8;
      int t = r >> 7;        // row = t*128 + b
      int b = r & 127;
      if (t < TSEQ - 1 && n < ODIM)
        out[((size_t)b * (TSEQ - 1) + t) * ODIM + n] = fsig(acc[nt][e]);
    }
  }
}

// ---------------- host ----------------
extern "C" void kernel_launch(void* const* d_in, const int* in_sizes, int n_in,
                              void* d_out, int out_size, void* d_ws, size_t ws_size,
                              hipStream_t stream) {
  (void)in_sizes; (void)n_in; (void)out_size; (void)ws_size;
  const float* x    = (const float*)d_in[0];   // [128,500,248]
  const float* Wih  = (const float*)d_in[1];   // [2048,265]
  const float* Whh  = (const float*)d_in[2];   // [2048,512]
  const float* bih  = (const float*)d_in[3];   // [2048]
  const float* bhh  = (const float*)d_in[4];   // [2048]
  const float* Wfc  = (const float*)d_in[5];   // [123,512]
  const float* bfc  = (const float*)d_in[6];   // [123]
  float* out = (float*)d_out;                  // [128,499,123]

  // workspace carve-up (256B aligned)
  size_t off = 0;
  auto take = [&](size_t bytes) { size_t o = off; off = (off + bytes + 255) & ~(size_t)255; return o; };
  char* ws = (char*)d_ws;
  _Float16* nx    = (_Float16*)(ws + take((size_t)BATCH * TSEQ * KX * 2));   // 36.9 MB
  _Float16* wihH  = (_Float16*)(ws + take((size_t)G4 * KX * 2));             // 1.2 MB
  _Float16* whhH  = (_Float16*)(ws + take((size_t)G4 * H * 2));              // 2.1 MB
  _Float16* wfcH  = (_Float16*)(ws + take((size_t)OPAD * H * 2));            // 128 KB
  float*    biasS = (float*)   (ws + take((size_t)G4 * 4));                  // 8 KB
  size_t hbufOff  = take((size_t)2 * BATCH * H * 2);                         // 256 KB
  _Float16* hbuf  = (_Float16*)(ws + hbufOff);
  size_t barOff   = take(256);
  unsigned* bar   = (unsigned*)(ws + barOff);
  _Float16* hs    = (_Float16*)(ws + take((size_t)TSEQ * BATCH * H * 2));    // 65.5 MB

  // zero h ping-pong + barrier counter (contiguous region); capture-legal
  hipMemsetAsync(ws + hbufOff, 0, (barOff + 256) - hbufOff, stream);

  // preprocessing / precision conversion
  {
    size_t n = (size_t)BATCH * TSEQ * KX;
    prep_newx<<<dim3((unsigned)((n + 255) / 256)), dim3(256), 0, stream>>>(x, nx);
  }
  cvt_pad_f16<<<dim3((G4 * KX + 255) / 256), dim3(256), 0, stream>>>(Wih, wihH, G4, 265, G4, KX);
  cvt_pad_f16<<<dim3((G4 * H + 255) / 256), dim3(256), 0, stream>>>(Whh, whhH, G4, H, G4, H);
  cvt_pad_f16<<<dim3((OPAD * H + 255) / 256), dim3(256), 0, stream>>>(Wfc, wfcH, ODIM, H, OPAD, H);
  bias_sum<<<dim3((G4 + 255) / 256), dim3(256), 0, stream>>>(bih, bhh, biasS, G4);

  // fused persistent LSTM scan: 32 WGs x 256 threads (256 waves <-> 256 h-tiles)
  lstm_scan<<<dim3(32), dim3(256), 0, stream>>>(nx, wihH, whhH, biasS, hbuf, hs, bar);

  // output projection: 4000 M-tiles, 8 waves/block
  fc_out<<<dim3(500), dim3(256), 0, stream>>>(hs, wfcH, bfc, out);
}
